// TemporalAttention_79585743995191
// MI455X (gfx1250) — matrix operand order repaired and checked by reading, not compiled, not run on hardware
//
#include <hip/hip_runtime.h>
#include <hip/hip_bf16.h>

// ---------------------------------------------------------------------------
// TemporalAttention for MI455X (gfx1250, wave32, WMMA)
//   B=4, T=2048, D=512, HIDDEN=1024, H=16, Dh=64
//   d_out = [ out (B*T*1024 f32) | attn (B*H*T*T f32) ]
// ---------------------------------------------------------------------------

typedef __attribute__((ext_vector_type(16))) __bf16 v16bf;
typedef __attribute__((ext_vector_type(8)))  float  v8f;
typedef __attribute__((ext_vector_type(2)))  __bf16 bf16x2;

#define WMMA_BF16(a, b, c) \
    __builtin_amdgcn_wmma_f32_16x16x32_bf16(false, (a), false, (b), (short)0, (c), false, false)

#define TA_B  4
#define TA_T  2048
#define TA_D  512
#define TA_HD 1024
#define TA_H  16
#define TA_DH 64

// ---------------------------------------------------------------------------
// 1) f32 -> bf16 conversion (+ weight transposes so B-fragments load as
//    contiguous K-pairs).
// ---------------------------------------------------------------------------
__global__ __launch_bounds__(256) void ta_convert(
    const float* __restrict__ X,  const float* __restrict__ Wq,
    const float* __restrict__ Wk, const float* __restrict__ Wv,
    const float* __restrict__ Wo,
    __bf16* __restrict__ Xbf,  __bf16* __restrict__ WqT,
    __bf16* __restrict__ WkT,  __bf16* __restrict__ WvT,
    __bf16* __restrict__ WoT)
{
    size_t idx = (size_t)blockIdx.x * 256 + threadIdx.x;
    const size_t NX  = (size_t)TA_B * TA_T * TA_D;    // 4,194,304
    const size_t NW  = (size_t)TA_D * TA_HD;          //   524,288
    const size_t NWO = (size_t)TA_HD * TA_HD;         // 1,048,576
    if (idx < NX) { Xbf[idx] = (__bf16)X[idx]; return; }
    idx -= NX;
    if (idx < NW) { size_t n = idx >> 9, k = idx & 511;
                    WqT[idx] = (__bf16)Wq[k * TA_HD + n]; return; }
    idx -= NW;
    if (idx < NW) { size_t n = idx >> 9, k = idx & 511;
                    WkT[idx] = (__bf16)Wk[k * TA_HD + n]; return; }
    idx -= NW;
    if (idx < NW) { size_t n = idx >> 9, k = idx & 511;
                    WvT[idx] = (__bf16)Wv[k * TA_HD + n]; return; }
    idx -= NW;
    if (idx < NWO) { size_t n = idx >> 10, k = idx & 1023;
                     WoT[idx] = (__bf16)Wo[k * TA_HD + n]; }
}

// ---------------------------------------------------------------------------
// 2) bf16 WMMA GEMM:  C[M,N] = A[M,K] * BT[N,K]^T + bias
//    One wave computes a 16x64 output strip (4 N-tiles), K-loop step 32.
//    Fragments use contiguous k = 16*half + s in both A (16x32) and B (32x16).
// ---------------------------------------------------------------------------
__global__ __launch_bounds__(128) void ta_gemm_bf16(
    const __bf16* __restrict__ A, const __bf16* __restrict__ BT,
    const float* __restrict__ bias,
    __bf16* __restrict__ outBf, float* __restrict__ outF,
    int M, int N, int K)
{
    const int lane   = threadIdx.x & 31;
    const int wave   = threadIdx.x >> 5;
    const int waveId = blockIdx.x * 4 + wave;
    const int strips = N >> 6;
    const int mrow   = waveId / strips;
    const int ns     = waveId % strips;
    if (mrow * 16 >= M) return;                 // wave-uniform guard
    const int row0 = mrow << 4;
    const int n0   = ns << 6;
    const int m    = lane & 15;
    const int half = lane >> 4;

    v8f acc[4];
    #pragma unroll
    for (int nt = 0; nt < 4; ++nt)
        #pragma unroll
        for (int i = 0; i < 8; ++i) acc[nt][i] = 0.0f;

    const __bf16* Arow = A + (size_t)(row0 + m) * K;
    for (int k0 = 0; k0 < K; k0 += 32) {
        // A fragment (16x32 bf16): lane m = row; VGPR j holds K pair
        v16bf a;
        #pragma unroll
        for (int j = 0; j < 8; ++j) {
            const int kb = half * 16 + j * 2;
            bf16x2 p = *(const bf16x2*)(Arow + k0 + kb);
            a[2 * j] = p.x; a[2 * j + 1] = p.y;
        }
        __builtin_prefetch(Arow + k0 + 64, 0, 0);   // global_prefetch_b8
        #pragma unroll
        for (int nt = 0; nt < 4; ++nt) {
            // B fragment (32x16): lane holds column n, K pairs contiguous in BT
            const __bf16* Brow =
                BT + (size_t)(n0 + nt * 16 + m) * K + k0 + half * 16;
            v16bf b;
            #pragma unroll
            for (int j = 0; j < 8; ++j) {
                bf16x2 p = *(const bf16x2*)(Brow + 2 * j);
                b[2 * j] = p.x; b[2 * j + 1] = p.y;
            }
            acc[nt] = WMMA_BF16(a, b, acc[nt]);
        }
    }

    // D layout: lane -> (row = r + 8*half, col = lane&15)
    #pragma unroll
    for (int nt = 0; nt < 4; ++nt)
        #pragma unroll
        for (int r = 0; r < 8; ++r) {
            const int n   = n0 + nt * 16 + m;
            const int row = row0 + r + 8 * half;
            const float v = acc[nt][r] + bias[n];
            if (outBf) outBf[(size_t)row * N + n] = (__bf16)v;
            else       outF [(size_t)row * N + n] = v;
        }
}

// ---------------------------------------------------------------------------
// 3) Attention: one workgroup (8 waves) per (b, h, 16-query tile).
//    Wave w owns keys [w*256, w*256+256). Scores recomputed from register-
//    resident Q fragments in 3 passes (max, sum, emit+PV) to avoid spilling
//    raw scores to HBM; K/V head slices (256 KB) stay L2-resident.
// ---------------------------------------------------------------------------
__global__ __launch_bounds__(256) void ta_attention(
    const __bf16* __restrict__ Q, const __bf16* __restrict__ Kc,
    const __bf16* __restrict__ V, const int* __restrict__ maskp,
    float* __restrict__ attn, __bf16* __restrict__ Cbf)
{
    const int bid  = blockIdx.x;
    const int qt   = bid & 127;
    const int h    = (bid >> 7) & 15;
    const int b    = bid >> 11;
    const int lane = threadIdx.x & 31;
    const int wave = threadIdx.x >> 5;
    const int m    = lane & 15;
    const int half = lane >> 4;
    const int q0   = qt << 4;

    __shared__ float  redbuf[128];
    __shared__ float  gmax[16];
    __shared__ float  gsum[16];
    __shared__ float  ctx[1024];
    __shared__ __bf16 pst[8][512];   // per-wave 16x32 bf16 P staging

    for (int i = threadIdx.x; i < 1024; i += 256) ctx[i] = 0.0f;

    // Q A-fragments: two 16x32 chunks covering Dh=64
    v16bf qa[2];
    #pragma unroll
    for (int ks = 0; ks < 2; ++ks) {
        const __bf16* Qp =
            Q + (size_t)(b * TA_T + q0 + m) * TA_HD + h * TA_DH + ks * 32;
        #pragma unroll
        for (int j = 0; j < 8; ++j) {
            const int kb = half * 16 + j * 2;
            bf16x2 p = *(const bf16x2*)(Qp + kb);
            qa[ks][2 * j] = p.x; qa[ks][2 * j + 1] = p.y;
        }
    }

    const float scale = 0.125f;      // 1/sqrt(64)

    auto score_tile = [&](int kt) -> v8f {
        v8f s;
        #pragma unroll
        for (int i = 0; i < 8; ++i) s[i] = 0.0f;
        #pragma unroll
        for (int ks = 0; ks < 2; ++ks) {
            // B fragment = K^T: lane holds key n = kt*16+m, dims contiguous
            const __bf16* Kp = Kc + (size_t)(b * TA_T + kt * 16 + m) * TA_HD +
                               h * TA_DH + ks * 32 + half * 16;
            v16bf bb;
            #pragma unroll
            for (int j = 0; j < 8; ++j) {
                bf16x2 p = *(const bf16x2*)(Kp + 2 * j);
                bb[2 * j] = p.x; bb[2 * j + 1] = p.y;
            }
            s = WMMA_BF16(qa[ks], bb, s);
        }
        return s;
    };

    // ---- Pass A: row maxima ----
    float rmax[8];
    #pragma unroll
    for (int r = 0; r < 8; ++r) rmax[r] = -3.0e38f;
    for (int t = 0; t < 16; ++t) {
        const int kt = wave * 16 + t;
        v8f s = score_tile(kt);
        const int   key  = kt * 16 + m;
        const float mval = (maskp[b * TA_T + key] != 0) ? 0.0f : -1.0e30f;
        #pragma unroll
        for (int r = 0; r < 8; ++r)
            rmax[r] = fmaxf(rmax[r], s[r] * scale + mval);
    }
    for (int off = 1; off < 16; off <<= 1)
        #pragma unroll
        for (int r = 0; r < 8; ++r)
            rmax[r] = fmaxf(rmax[r], __shfl_xor(rmax[r], off));
    if (m == 0)
        #pragma unroll
        for (int r = 0; r < 8; ++r) redbuf[wave * 16 + half * 8 + r] = rmax[r];
    __syncthreads();
    if (threadIdx.x < 16) {
        float v = -3.0e38f;
        for (int w = 0; w < 8; ++w) v = fmaxf(v, redbuf[w * 16 + threadIdx.x]);
        gmax[threadIdx.x] = v;
    }
    __syncthreads();

    float rowm[8];
    #pragma unroll
    for (int r = 0; r < 8; ++r) rowm[r] = gmax[half * 8 + r];

    // ---- Pass B: row sums of exp ----
    float rsum[8];
    #pragma unroll
    for (int r = 0; r < 8; ++r) rsum[r] = 0.0f;
    for (int t = 0; t < 16; ++t) {
        const int kt = wave * 16 + t;
        v8f s = score_tile(kt);
        const int   key  = kt * 16 + m;
        const float mval = (maskp[b * TA_T + key] != 0) ? 0.0f : -1.0e30f;
        #pragma unroll
        for (int r = 0; r < 8; ++r)
            rsum[r] += __expf(s[r] * scale + mval - rowm[r]);
    }
    for (int off = 1; off < 16; off <<= 1)
        #pragma unroll
        for (int r = 0; r < 8; ++r) rsum[r] += __shfl_xor(rsum[r], off);
    if (m == 0)
        #pragma unroll
        for (int r = 0; r < 8; ++r) redbuf[wave * 16 + half * 8 + r] = rsum[r];
    __syncthreads();
    if (threadIdx.x < 16) {
        float v = 0.0f;
        for (int w = 0; w < 8; ++w) v += redbuf[w * 16 + threadIdx.x];
        gsum[threadIdx.x] = v;
    }
    __syncthreads();

    float rinv[8];
    #pragma unroll
    for (int r = 0; r < 8; ++r) rinv[r] = 1.0f / gsum[half * 8 + r];

    // ---- Pass C: emit attn, accumulate context = P * V via WMMA ----
    v8f acc[4];
    #pragma unroll
    for (int nt = 0; nt < 4; ++nt)
        #pragma unroll
        for (int i = 0; i < 8; ++i) acc[nt][i] = 0.0f;

    float* attnBase = attn + ((size_t)(b * TA_H + h) * TA_T + q0) * TA_T;
    for (int t = 0; t < 16; ++t) {
        const int kt = wave * 16 + t;
        v8f s = score_tile(kt);
        const int   key  = kt * 16 + m;
        const float mval = (maskp[b * TA_T + key] != 0) ? 0.0f : -1.0e30f;
        #pragma unroll
        for (int r = 0; r < 8; ++r) {
            const float p = __expf(s[r] * scale + mval - rowm[r]) * rinv[r];
            attnBase[(size_t)(half * 8 + r) * TA_T + key] = p;
            pst[wave][(half * 8 + r) * 32 + (t & 1) * 16 + m] = (__bf16)p;
        }
        if (t & 1) {
            asm volatile("s_wait_dscnt 0" ::: "memory");  // wave-local LDS sync
            // Re-read P in A-matrix layout
            v16bf pa;
            #pragma unroll
            for (int j = 0; j < 8; ++j) {
                const int kb = half * 16 + j * 2;
                pa[2 * j]     = pst[wave][m * 32 + kb];
                pa[2 * j + 1] = pst[wave][m * 32 + kb + 1];
            }
            const int keyb = (kt - 1) * 16;   // 32 keys in this chunk
            #pragma unroll
            for (int nt = 0; nt < 4; ++nt) {
                // B fragment = V chunk 32(keys) x 16(dims)
                const __bf16* Vp = V +
                    (size_t)(b * TA_T + keyb + half * 16) * TA_HD +
                    h * TA_DH + nt * 16 + m;
                v16bf vb;
                #pragma unroll
                for (int j = 0; j < 8; ++j) {
                    vb[2 * j]     = Vp[(size_t)(2 * j)     * TA_HD];
                    vb[2 * j + 1] = Vp[(size_t)(2 * j + 1) * TA_HD];
                }
                acc[nt] = WMMA_BF16(pa, vb, acc[nt]);
            }
        }
    }

    // Merge per-wave partial contexts (ds_add_f32)
    #pragma unroll
    for (int nt = 0; nt < 4; ++nt)
        #pragma unroll
        for (int r = 0; r < 8; ++r)
            atomicAdd(&ctx[(half * 8 + r) * 64 + nt * 16 + m], acc[nt][r]);
    __syncthreads();

    for (int i = threadIdx.x; i < 1024; i += 256) {
        const int row = i >> 6, col = i & 63;
        Cbf[(size_t)(b * TA_T + q0 + row) * TA_HD + h * TA_DH + col] =
            (__bf16)ctx[i];
    }
}

// ---------------------------------------------------------------------------
// Launch
// ---------------------------------------------------------------------------
extern "C" void kernel_launch(void* const* d_in, const int* in_sizes, int n_in,
                              void* d_out, int out_size, void* d_ws, size_t ws_size,
                              hipStream_t stream)
{
    (void)in_sizes; (void)n_in; (void)out_size; (void)ws_size;

    const float* X    = (const float*)d_in[0];
    const int*   mask = (const int*)d_in[1];
    const float* Wq   = (const float*)d_in[2];
    const float* bq   = (const float*)d_in[3];
    const float* Wk   = (const float*)d_in[4];
    const float* bk   = (const float*)d_in[5];
    const float* Wv   = (const float*)d_in[6];
    const float* bv   = (const float*)d_in[7];
    const float* Wo   = (const float*)d_in[8];
    const float* bo   = (const float*)d_in[9];

    // Workspace layout (bf16 elements)
    const size_t N_X   = (size_t)TA_B * TA_T * TA_D;   // 4,194,304
    const size_t N_W   = (size_t)TA_D * TA_HD;         //   524,288
    const size_t N_WO  = (size_t)TA_HD * TA_HD;        // 1,048,576
    const size_t N_QKV = (size_t)TA_B * TA_T * TA_HD;  // 8,388,608

    __bf16* base = (__bf16*)d_ws;
    __bf16* Xbf = base;
    __bf16* WqT = Xbf + N_X;
    __bf16* WkT = WqT + N_W;
    __bf16* WvT = WkT + N_W;
    __bf16* WoT = WvT + N_W;
    __bf16* Qbf = WoT + N_WO;
    __bf16* Kbf = Qbf + N_QKV;
    __bf16* Vbf = Kbf + N_QKV;
    __bf16* Cbf = Vbf + N_QKV;                         // total ~81 MB

    float* out  = (float*)d_out;
    float* attn = out + N_QKV;                         // out is f32 B*T*1024

    // 1) Conversions / transposes
    const size_t convTotal = N_X + 3 * N_W + N_WO;     // 6,815,744
    ta_convert<<<(unsigned)((convTotal + 255) / 256), 256, 0, stream>>>(
        X, Wq, Wk, Wv, Wo, Xbf, WqT, WkT, WvT, WoT);

    // 2) Q/K/V projections: M=8192, N=1024, K=512 -> 8192 waves / 4 per block
    ta_gemm_bf16<<<2048, 128, 0, stream>>>(Xbf, WqT, bq, Qbf, nullptr,
                                           TA_B * TA_T, TA_HD, TA_D);
    ta_gemm_bf16<<<2048, 128, 0, stream>>>(Xbf, WkT, bk, Kbf, nullptr,
                                           TA_B * TA_T, TA_HD, TA_D);
    ta_gemm_bf16<<<2048, 128, 0, stream>>>(Xbf, WvT, bv, Vbf, nullptr,
                                           TA_B * TA_T, TA_HD, TA_D);

    // 3) Attention: B*H*(T/16) = 8192 blocks of 256 threads (8 waves)
    ta_attention<<<TA_B * TA_H * (TA_T / 16), 256, 0, stream>>>(
        Qbf, Kbf, Vbf, mask, attn, Cbf);

    // 4) Output projection: M=8192, N=1024, K=1024, f32 out
    ta_gemm_bf16<<<2048, 128, 0, stream>>>(Cbf, WoT, bo, nullptr, out,
                                           TA_B * TA_T, TA_HD, TA_HD);
}
